// DGCN1_71863392796964
// MI455X (gfx1250) — matrix operand order repaired and checked by reading, not compiled, not run on hardware
//
#include <hip/hip_runtime.h>
#include <math.h>

typedef __attribute__((ext_vector_type(16))) __bf16 v16bf;
typedef __attribute__((ext_vector_type(8)))  __bf16 v8bf;
typedef __attribute__((ext_vector_type(8)))  float  v8f;

constexpr int Bq = 8, Cc = 128, Nn = 1024, Tt = 32;
constexpr int NT = Nn * Tt;           // 32768 positions per (b)
constexpr int K2C = 2 * Cc;           // 256
constexpr int TOPK = 819;             // int(1024*0.8)
#define SCALE_C 0.08838834764831845f  // 1/sqrt(128)
#define NEGINF  (-1e12f)

// ---------------- WMMA helpers (CDNA5 bf16 16x16x32, fp32 accum) ----------------

__device__ __forceinline__ v16bf cat8(v8bf lo, v8bf hi) {
  v16bf r;
#pragma unroll
  for (int i = 0; i < 8; i++) { r[i] = lo[i]; r[i + 8] = hi[i]; }
  return r;
}

// A-tile 16(M)x32(K), row-major source with row stride `stride` (elements).
// Layout: lane<16 holds row M=lane, K = k0..k0+15;
//         lane>=16 holds row M=lane-16, K = k0+16..k0+31 (matches B).
__device__ __forceinline__ v16bf loadA_bf16(const __bf16* base, int stride, int lane) {
  const __bf16* p = base + (lane & 15) * stride + ((lane >> 4) << 4);
  v8bf lo = *(const v8bf*)p;
  v8bf hi = *(const v8bf*)(p + 8);
  return cat8(lo, hi);
}

__device__ __forceinline__ v16bf loadA_f32(const float* base, int stride, int lane) {
  const float* p = base + (lane & 15) * stride + ((lane >> 4) << 4);
  v16bf r;
#pragma unroll
  for (int i = 0; i < 8; i++) { r[i] = (__bf16)p[i]; r[i + 8] = (__bf16)p[i + 8]; }
  return r;
}

// B-tile 32(K)x16(Ncol) loaded from B^T (col-major: [col][k] contiguous, stride elems).
// Layout used here (matches A): lane holds column lane%16; K-range = (lane/16)*16 .. +15 (contiguous).
__device__ __forceinline__ v16bf loadB_bf16(const __bf16* base, int stride, int lane) {
  const __bf16* p = base + (lane & 15) * stride + ((lane >> 4) << 4);
  return *(const v16bf*)p;  // 32B aligned by construction
}

__device__ __forceinline__ v8f wmma_bf16(v16bf a, v16bf b, v8f c) {
  return __builtin_amdgcn_wmma_f32_16x16x32_bf16(false, a, false, b, (short)0, c,
                                                 false, false);
}

// D/C tile: lane holds col lane%16; VGPR v holds row v + (lane/16)*8.
__device__ __forceinline__ void storeD_bf16(__bf16* base, int stride, int lane, v8f acc) {
  int col = lane & 15, rofs = (lane >> 4) * 8;
#pragma unroll
  for (int v = 0; v < 8; v++) base[(size_t)(rofs + v) * stride + col] = (__bf16)acc[v];
}

// ---------------- K1: x[b,c,n,t] fp32 -> xT[b,p,c] bf16 (p = n*T+t) ----------------
__global__ void k_x_to_xT(const float* __restrict__ x, __bf16* __restrict__ xT) {
  __shared__ float tile[32][33];
  int b = blockIdx.z, c0 = blockIdx.y * 32, p0 = blockIdx.x * 32, tid = threadIdx.x;
#pragma unroll
  for (int i = 0; i < 4; i++) {
    int idx = tid + i * 256, r = idx >> 5, col = idx & 31;
    tile[r][col] = x[((size_t)(b * Cc + c0 + r)) * NT + p0 + col];
  }
  __syncthreads();
#pragma unroll
  for (int i = 0; i < 4; i++) {
    int idx = tid + i * 256, r = idx >> 5, col = idx & 31;  // r: p-local, col: c-local
    xT[((size_t)b * NT + p0 + r) * Cc + c0 + col] = (__bf16)tile[col][r];
  }
}

// ---------------- K2: conv GEMM  yT[b,c,t,n] = bf16(conv_w @ x + conv_b) ----------------
// grid (N/64, T, B), block 256 (8 waves). Wave w owns d-rows [16w,16w+16); 4 n-subtiles.
__global__ void k_conv(const float* __restrict__ convw, const float* __restrict__ convb,
                       const __bf16* __restrict__ xT, __bf16* __restrict__ yT) {
  int b = blockIdx.z, t = blockIdx.y, n0 = blockIdx.x * 64;
  int lane = threadIdx.x & 31, wave = threadIdx.x >> 5;
  int d0 = wave * 16;
  v8f acc[4] = {};
  for (int k0 = 0; k0 < Cc; k0 += 32) {
    v16bf a = loadA_f32(convw + d0 * Cc + k0, Cc, lane);
#pragma unroll
    for (int s = 0; s < 4; s++) {
      int n = n0 + s * 16 + (lane & 15);
      const __bf16* bp = xT + ((size_t)b * NT + (size_t)n * Tt + t) * Cc + k0 + ((lane >> 4) << 4);
      v16bf bb = cat8(*(const v8bf*)bp, *(const v8bf*)(bp + 8));
      acc[s] = wmma_bf16(a, bb, acc[s]);
    }
  }
  int col = lane & 15, rofs = (lane >> 4) * 8;
#pragma unroll
  for (int s = 0; s < 4; s++)
#pragma unroll
    for (int v = 0; v < 8; v++) {
      int d = d0 + rofs + v;
      float val = acc[s][v] + convb[d];
      yT[(((size_t)(b * Cc + d)) * Tt + t) * Nn + (n0 + s * 16 + col)] = (__bf16)val;
    }
}

// ---------------- K3: ysum[b,n,c] = sum_t yT[b,c,t,n] ----------------
__global__ void k_ysum(const __bf16* __restrict__ yT, float* __restrict__ ysum) {
  int b = blockIdx.z, c = blockIdx.y * 8 + (threadIdx.x >> 5),
      n = blockIdx.x * 32 + (threadIdx.x & 31);
  const __bf16* p = yT + ((size_t)(b * Cc + c) * Tt) * Nn + n;
  float s = 0.f;
#pragma unroll
  for (int t = 0; t < Tt; t++) s += (float)p[(size_t)t * Nn];
  ysum[((size_t)b * Nn + n) * Cc + c] = s;
}

// ---------------- K4: hs = ysum @ W ; Wh1/Wh2 dot a_vec ----------------
__global__ void k_hs(const float* __restrict__ ysum, const float* __restrict__ W,
                     const float* __restrict__ avec, float* __restrict__ hs,
                     float* __restrict__ wh1, float* __restrict__ wh2) {
  __shared__ float ys[Cc];
  __shared__ float red[Cc];
  int bn = blockIdx.x, c = threadIdx.x;
  ys[c] = ysum[(size_t)bn * Cc + c];
  __syncthreads();
  float acc = 0.f;
  for (int cc = 0; cc < Cc; cc++) acc = fmaf(ys[cc], W[cc * Cc + c], acc);
  hs[(size_t)bn * Cc + c] = acc;
  red[c] = acc * avec[c];
  __syncthreads();
  for (int s = 64; s > 0; s >>= 1) { if (c < s) red[c] += red[c + s]; __syncthreads(); }
  if (c == 0) wh1[bn] = red[0];
  __syncthreads();
  red[c] = acc * avec[Cc + c];
  __syncthreads();
  for (int s = 64; s > 0; s >>= 1) { if (c < s) red[c] += red[c + s]; __syncthreads(); }
  if (c == 0) wh2[bn] = red[0];
}

// ---------------- K5: per-row adjacency + exact top-k + masked softmax att ----------------
// grid (N, B), block 256. Writes attB[b,n,m] bf16.
__global__ void k_att(const float* __restrict__ hs, const float* __restrict__ memv,
                      const float* __restrict__ cwa, const float* __restrict__ cw,
                      const float* __restrict__ wh1, const float* __restrict__ wh2,
                      __bf16* __restrict__ attB) {
  __shared__ float hsn[Cc];
  __shared__ float s1[Nn];
  __shared__ float s2[Nn];
  __shared__ float red[256];
  int b = blockIdx.y, n = blockIdx.x, tid = threadIdx.x;
  if (tid < Cc) hsn[tid] = hs[((size_t)b * Nn + n) * Cc + tid];
  __syncthreads();
  for (int m = tid; m < Nn; m += 256) {
    const float4* mr = (const float4*)(memv + (size_t)m * Cc);
    const float4* hr = (const float4*)(hs + ((size_t)b * Nn + m) * Cc);
    float d1 = 0.f, d2 = 0.f;
#pragma unroll 4
    for (int q = 0; q < Cc / 4; q++) {
      float4 mv = mr[q], hv = hr[q];
      float h0 = hsn[4 * q], h1 = hsn[4 * q + 1], h2 = hsn[4 * q + 2], h3 = hsn[4 * q + 3];
      d1 += h0 * mv.x + h1 * mv.y + h2 * mv.z + h3 * mv.w;
      d2 += h0 * hv.x + h1 * hv.y + h2 * hv.z + h3 * hv.w;
    }
    d1 *= SCALE_C; d2 *= SCALE_C;
    s1[m] = d1 > 0.f ? d1 : 0.f;
    s2[m] = d2 > 0.f ? d2 : 0.f;
  }
  __syncthreads();
  // softmax(s1) -> adj1 in place
  float lm = -INFINITY;
  for (int m = tid; m < Nn; m += 256) lm = fmaxf(lm, s1[m]);
  red[tid] = lm; __syncthreads();
  for (int s = 128; s > 0; s >>= 1) { if (tid < s) red[tid] = fmaxf(red[tid], red[tid + s]); __syncthreads(); }
  float mx = red[0]; __syncthreads();
  float ls = 0.f;
  for (int m = tid; m < Nn; m += 256) { float e = expf(s1[m] - mx); s1[m] = e; ls += e; }
  red[tid] = ls; __syncthreads();
  for (int s = 128; s > 0; s >>= 1) { if (tid < s) red[tid] += red[tid + s]; __syncthreads(); }
  float inv1 = 1.f / red[0]; __syncthreads();
  // softmax(s2) -> adj2 in place
  lm = -INFINITY;
  for (int m = tid; m < Nn; m += 256) lm = fmaxf(lm, s2[m]);
  red[tid] = lm; __syncthreads();
  for (int s = 128; s > 0; s >>= 1) { if (tid < s) red[tid] = fmaxf(red[tid], red[tid + s]); __syncthreads(); }
  mx = red[0]; __syncthreads();
  ls = 0.f;
  for (int m = tid; m < Nn; m += 256) { float e = expf(s2[m] - mx); s2[m] = e; ls += e; }
  red[tid] = ls; __syncthreads();
  for (int s = 128; s > 0; s >>= 1) { if (tid < s) red[tid] += red[tid + s]; __syncthreads(); }
  float inv2 = 1.f / red[0]; __syncthreads();
  // f = relu(cwa*adj1 + cw*adj2) -> s1 (each m owned by one thread; safe in-place)
  for (int m = tid; m < Nn; m += 256) {
    float v = cwa[(size_t)n * Nn + m] * (s1[m] * inv1) + cw[(size_t)n * Nn + m] * (s2[m] * inv2);
    s1[m] = v > 0.f ? v : 0.f;
  }
  __syncthreads();
  // exact top-k rank (matches jax.lax.top_k tie-break: lower index wins) + leaky att
  float w1 = wh1[(size_t)b * Nn + n];
  for (int m = tid; m < Nn; m += 256) {
    float fm = s1[m];
    int cnt = 0;
    for (int j = 0; j < Nn; j++) {
      float fj = s1[j];
      cnt += (fj > fm) || (fj == fm && j < m);
    }
    float e;
    if (cnt < TOPK) {
      float ev = w1 + wh2[(size_t)b * Nn + m];
      e = ev >= 0.f ? ev : 0.01f * ev;          // leaky_relu(e, 0.01)
    } else e = NEGINF;
    s2[m] = e;
  }
  __syncthreads();
  // final softmax over masked row -> attB
  lm = -INFINITY;
  for (int m = tid; m < Nn; m += 256) lm = fmaxf(lm, s2[m]);
  red[tid] = lm; __syncthreads();
  for (int s = 128; s > 0; s >>= 1) { if (tid < s) red[tid] = fmaxf(red[tid], red[tid + s]); __syncthreads(); }
  mx = red[0]; __syncthreads();
  ls = 0.f;
  for (int m = tid; m < Nn; m += 256) { float e = expf(s2[m] - mx); s2[m] = e; ls += e; }
  red[tid] = ls; __syncthreads();
  for (int s = 128; s > 0; s >>= 1) { if (tid < s) red[tid] += red[tid + s]; __syncthreads(); }
  float inv = 1.f / red[0]; __syncthreads();
  for (int m = tid; m < Nn; m += 256)
    attB[((size_t)b * Nn + n) * Nn + m] = (__bf16)(s2[m] * inv);
}

// ---------------- K6: attT[b,m,n] = attB[b,n,m] (bf16 tiled transpose) ----------------
__global__ void k_attT(const __bf16* __restrict__ attB, __bf16* __restrict__ attT) {
  __shared__ __bf16 tile[32][33];
  int b = blockIdx.z, n0 = blockIdx.y * 32, m0 = blockIdx.x * 32, tid = threadIdx.x;
#pragma unroll
  for (int i = 0; i < 4; i++) {
    int idx = tid + i * 256, r = idx >> 5, col = idx & 31;
    tile[r][col] = attB[((size_t)b * Nn + n0 + r) * Nn + m0 + col];
  }
  __syncthreads();
#pragma unroll
  for (int i = 0; i < 4; i++) {
    int idx = tid + i * 256, r = idx >> 5, col = idx & 31;  // r: m-local, col: n-local
    attT[((size_t)b * Nn + m0 + r) * Nn + n0 + col] = tile[col][r];
  }
}

// ---------------- K7/K8: diffusion GEMM  O[r,m] = sum_n A[r,n] * att[n,m] ----------------
// A: [4096 x 1024] bf16 row-major per batch; B^T = attT.
// grid (16, 16, B), block 256 (8 waves). Wave tile: 32 rows x 64 cols
// (2 A-tiles x 4 B-tiles, 8 accumulators) -> 6 tile-loads per 8 WMMAs.
__global__ void k_gemm_att(const __bf16* __restrict__ Ag, const __bf16* __restrict__ Bt,
                           __bf16* __restrict__ Og) {
  size_t aoff = (size_t)blockIdx.z * 4096 * Nn;
  size_t boff = (size_t)blockIdx.z * Nn * Nn;
  const __bf16* A = Ag + aoff;
  const __bf16* Bm = Bt + boff;
  __bf16* O = Og + aoff;
  int lane = threadIdx.x & 31, wave = threadIdx.x >> 5;
  int r0 = blockIdx.y * 256 + wave * 32;
  int c0 = blockIdx.x * 64;
  v8f acc0[4] = {};
  v8f acc1[4] = {};
  for (int k0 = 0; k0 < Nn; k0 += 32) {
    v16bf a0 = loadA_bf16(A + (size_t)r0 * Nn + k0, Nn, lane);
    v16bf a1 = loadA_bf16(A + (size_t)(r0 + 16) * Nn + k0, Nn, lane);
#pragma unroll
    for (int s = 0; s < 4; s++) {
      v16bf bb = loadB_bf16(Bm + (size_t)(c0 + s * 16) * Nn + k0, Nn, lane);
      acc0[s] = wmma_bf16(a0, bb, acc0[s]);
      acc1[s] = wmma_bf16(a1, bb, acc1[s]);
    }
  }
#pragma unroll
  for (int s = 0; s < 4; s++) {
    storeD_bf16(O + (size_t)r0 * Nn + c0 + s * 16, Nn, lane, acc0[s]);
    storeD_bf16(O + (size_t)(r0 + 16) * Nn + c0 + s * 16, Nn, lane, acc1[s]);
  }
}

// ---------------- K9: gcn GEMM  outT[b,d,t,n] = sum_{c2<256} gcn_w[d,c2]*xc[c2,t,n] ----------------
// grid (N/64, T, B), block 256. B tiles staged/transposed through LDS.
__global__ void k_xg(const float* __restrict__ gcnw, const __bf16* __restrict__ x1,
                     const __bf16* __restrict__ x2, __bf16* __restrict__ outT) {
  __shared__ __align__(16) __bf16 ldsB[64][40];
  int b = blockIdx.z, t = blockIdx.y, n0 = blockIdx.x * 64;
  int lane = threadIdx.x & 31, wave = threadIdx.x >> 5, tid = threadIdx.x;
  int d0 = wave * 16;
  v8f acc[4] = {};
  for (int k0 = 0; k0 < K2C; k0 += 32) {
    __syncthreads();
#pragma unroll
    for (int i = 0; i < 8; i++) {
      int idx = tid + i * 256;
      int col = idx & 63, kk = idx >> 6;
      int c2 = k0 + kk;
      const __bf16* src = (c2 < Cc) ? x1 : x2;
      int cc = c2 & (Cc - 1);
      ldsB[col][kk] = src[(((size_t)b * Cc + cc) * Tt + t) * Nn + n0 + col];
    }
    __syncthreads();
    v16bf a = loadA_f32(gcnw + d0 * K2C + k0, K2C, lane);
#pragma unroll
    for (int s = 0; s < 4; s++) {
      const __bf16* bp = &ldsB[s * 16 + (lane & 15)][(lane >> 4) * 16];
      v16bf bb = cat8(*(const v8bf*)bp, *(const v8bf*)(bp + 8));
      acc[s] = wmma_bf16(a, bb, acc[s]);
    }
  }
  int col = lane & 15, rofs = (lane >> 4) * 8;
#pragma unroll
  for (int s = 0; s < 4; s++)
#pragma unroll
    for (int v = 0; v < 8; v++) {
      int d = d0 + rofs + v;
      outT[(((size_t)(b * Cc + d)) * Tt + t) * Nn + (n0 + s * 16 + col)] = (__bf16)acc[s][v];
    }
}

// ---------------- K10: out[b,d,n,t] = (outT[b,d,t,n] + gcn_b[d]) * emb[d] + x[b,d,n,t] ----------------
__global__ void k_final(const __bf16* __restrict__ outT, const float* __restrict__ x,
                        const float* __restrict__ gcnb, const float* __restrict__ emb,
                        float* __restrict__ out) {
  __shared__ float tile[32][33];
  int b = blockIdx.z, d = blockIdx.y, n0 = blockIdx.x * 32, tid = threadIdx.x;
  float bias = gcnb[d], ee = emb[d];
#pragma unroll
  for (int i = 0; i < 4; i++) {
    int idx = tid + i * 256, tt = idx >> 5, col = idx & 31;  // row t, col n
    tile[tt][col] = (float)outT[(((size_t)(b * Cc + d)) * Tt + tt) * Nn + n0 + col];
  }
  __syncthreads();
#pragma unroll
  for (int i = 0; i < 4; i++) {
    int idx = tid + i * 256, r = idx >> 5, col = idx & 31;  // row n-local, col t
    size_t o = (((size_t)(b * Cc + d)) * Nn + n0 + r) * Tt + col;
    out[o] = (tile[col][r] + bias) * ee + x[o];
  }
}

// ---------------- launch ----------------
extern "C" void kernel_launch(void* const* d_in, const int* in_sizes, int n_in,
                              void* d_out, int out_size, void* d_ws, size_t ws_size,
                              hipStream_t stream) {
  (void)in_sizes; (void)n_in; (void)out_size; (void)ws_size;
  const float* x      = (const float*)d_in[0];
  const float* conv_w = (const float*)d_in[1];
  const float* conv_b = (const float*)d_in[2];
  const float* W      = (const float*)d_in[3];
  const float* a_vec  = (const float*)d_in[4];
  const float* memory = (const float*)d_in[5];
  const float* cwa    = (const float*)d_in[6];
  const float* cw     = (const float*)d_in[7];
  const float* gcn_w  = (const float*)d_in[8];
  const float* gcn_b  = (const float*)d_in[9];
  const float* emb    = (const float*)d_in[10];
  float* out = (float*)d_out;

  char* ws = (char*)d_ws;
  const size_t big = (size_t)Bq * Cc * NT * 2;  // 67,108,864 B (bf16 [B,C,T,N]-sized)
  __bf16* xT_x1 = (__bf16*)ws; ws += big;       // xT, later reused as x1
  __bf16* yT_x2 = (__bf16*)ws; ws += big;       // yT, later reused as x2
  __bf16* outT  = (__bf16*)ws; ws += big;
  __bf16* attB  = (__bf16*)ws; ws += (size_t)Bq * Nn * Nn * 2;
  __bf16* attT  = (__bf16*)ws; ws += (size_t)Bq * Nn * Nn * 2;
  float*  ysum  = (float*)ws;  ws += (size_t)Bq * Nn * Cc * 4;
  float*  hs    = (float*)ws;  ws += (size_t)Bq * Nn * Cc * 4;
  float*  wh1   = (float*)ws;  ws += (size_t)Bq * Nn * 4;
  float*  wh2   = (float*)ws;  ws += (size_t)Bq * Nn * 4;

  k_x_to_xT <<<dim3(NT / 32, Cc / 32, Bq), 256, 0, stream>>>(x, xT_x1);
  k_conv    <<<dim3(Nn / 64, Tt, Bq),      256, 0, stream>>>(conv_w, conv_b, xT_x1, yT_x2);
  k_ysum    <<<dim3(Nn / 32, Cc / 8, Bq),  256, 0, stream>>>(yT_x2, ysum);
  k_hs      <<<dim3(Bq * Nn),              128, 0, stream>>>(ysum, W, a_vec, hs, wh1, wh2);
  k_att     <<<dim3(Nn, Bq),               256, 0, stream>>>(hs, memory, cwa, cw, wh1, wh2, attB);
  k_attT    <<<dim3(Nn / 32, Nn / 32, Bq), 256, 0, stream>>>(attB, attT);
  k_gemm_att<<<dim3(16, 16, Bq),           256, 0, stream>>>(yT_x2, attT, xT_x1);  // x1
  k_gemm_att<<<dim3(16, 16, Bq),           256, 0, stream>>>(xT_x1, attT, yT_x2);  // x2
  k_xg      <<<dim3(Nn / 64, Tt, Bq),      256, 0, stream>>>(gcn_w, xT_x1, yT_x2, outT);
  k_final   <<<dim3(Nn / 32, Cc, Bq),      256, 0, stream>>>(outT, x, gcn_b, emb, out);
}
